// MTKGNN_79602923864320
// MI455X (gfx1250) — compile-verified, matrix operand order checked by reading, not run on hardware
//
#include <hip/hip_runtime.h>
#include <hip/hip_bf16.h>
#include <stdint.h>

#define NENTITY   50000
#define NRELATION 500
#define RANK      128
#define BATCH     64
#define ENT_TILE  128   // entities staged per block

#if __has_builtin(__builtin_amdgcn_global_load_async_to_lds_b128) && \
    __has_builtin(__builtin_amdgcn_s_wait_asynccnt)
#define USE_ASYNC 1
#else
#define USE_ASYNC 0
#endif

#define AS1 __attribute__((address_space(1)))
#define AS3 __attribute__((address_space(3)))

typedef int v4i_vec __attribute__((vector_size(16)));  // matches builtin param

typedef __attribute__((ext_vector_type(16))) _Float16 v16h;
typedef __attribute__((ext_vector_type(8)))  float    v8f;
typedef __attribute__((ext_vector_type(2)))  _Float16 h2;

union AFrag { v16h v; uint32_t u[8]; };

__device__ __forceinline__ uint32_t pack2h(float x, float y) {
  h2 r; r.x = (_Float16)x; r.y = (_Float16)y;
  return __builtin_bit_cast(uint32_t, r);
}

// packed |a-b| on two f16 lanes: v_pk_add_f16(neg) + v_and_b32
__device__ __forceinline__ uint32_t absdiff2(uint32_t pa, uint32_t pe) {
  h2 a = __builtin_bit_cast(h2, pa);
  h2 e = __builtin_bit_cast(h2, pe);
  h2 d = a - e;
  return __builtin_bit_cast(uint32_t, d) & 0x7FFF7FFFu;
}

__global__ __launch_bounds__(256) void transe_l1_wmma_kernel(
    const float* __restrict__ ent, const float* __restrict__ rel,
    const int* __restrict__ heads, const int* __restrict__ rels,
    float* __restrict__ out)
{
  __shared__ __align__(16) uint16_t s_ent[ENT_TILE * RANK];  // 32KB f16 tile
  __shared__ __align__(16) uint16_t s_pred[BATCH * RANK];    // 16KB f16 pred
#if USE_ASYNC
  __shared__ __align__(16) float s_stage[32 * RANK];         // 16KB f32 stage
#endif

  const int tid   = threadIdx.x;
  const int nbase = blockIdx.x * ENT_TILE;

#if USE_ASYNC
  // ---- stage entity tile via async copy (4 phases of 32 rows) ----
  #pragma unroll 1
  for (int ph = 0; ph < 4; ++ph) {
    const int rbase = ph * 32;
    // 32 rows * 32 16B-chunks = 1024 tasks, 4 async b128 issues per thread
    #pragma unroll
    for (int it = 0; it < 4; ++it) {
      int task = tid + it * 256;
      int rrow = task >> 5;
      int kq   = task & 31;
      int n = nbase + rbase + rrow;
      if (n >= NENTITY) n = NENTITY - 1;
      __builtin_amdgcn_global_load_async_to_lds_b128(
          (AS1 v4i_vec*)(ent + (size_t)n * RANK + kq * 4),
          (AS3 v4i_vec*)(s_stage + rrow * RANK + kq * 4), 0, 0);
    }
    if (ph == 0) {
      // ---- pred[b][k] = f16(ent[heads[b]][k] + rel[rels[b]][k]) ----
      // overlapped with phase-0 async latency
      #pragma unroll
      for (int it = 0; it < (BATCH * (RANK / 4)) / 256; ++it) {
        int task = tid + it * 256;
        int b  = task >> 5;
        int kq = task & 31;
        int h = heads[b];
        int r = rels[b];
        const float4 eh = *(const float4*)(ent + (size_t)h * RANK + kq * 4);
        const float4 rv = *(const float4*)(rel + (size_t)r * RANK + kq * 4);
        uint2 p;
        p.x = pack2h(eh.x + rv.x, eh.y + rv.y);
        p.y = pack2h(eh.z + rv.z, eh.w + rv.w);
        *(uint2*)(s_pred + b * RANK + kq * 4) = p;
      }
    }
    __builtin_amdgcn_s_wait_asynccnt(0);
    __syncthreads();
    // convert staged f32 -> f16 tile
    #pragma unroll
    for (int it = 0; it < 4; ++it) {
      int task = tid + it * 256;
      int rrow = task >> 5;
      int kq   = task & 31;
      const float4 e = *(const float4*)(s_stage + rrow * RANK + kq * 4);
      uint2 p;
      p.x = pack2h(e.x, e.y);
      p.y = pack2h(e.z, e.w);
      *(uint2*)(s_ent + (rbase + rrow) * RANK + kq * 4) = p;
    }
    __syncthreads();  // stage buffer reused next phase / tile complete
  }
#else
  // ---- fallback: synchronous staging ----
  #pragma unroll
  for (int it = 0; it < (BATCH * (RANK / 4)) / 256; ++it) {
    int task = tid + it * 256;
    int b  = task >> 5;
    int kq = task & 31;
    int h = heads[b];
    int r = rels[b];
    const float4 eh = *(const float4*)(ent + (size_t)h * RANK + kq * 4);
    const float4 rv = *(const float4*)(rel + (size_t)r * RANK + kq * 4);
    uint2 p;
    p.x = pack2h(eh.x + rv.x, eh.y + rv.y);
    p.y = pack2h(eh.z + rv.z, eh.w + rv.w);
    *(uint2*)(s_pred + b * RANK + kq * 4) = p;
  }
  #pragma unroll
  for (int it = 0; it < (ENT_TILE * (RANK / 4)) / 256; ++it) {
    int task = tid + it * 256;
    int rrow = task >> 5;
    int kq   = task & 31;
    int n = nbase + rrow;
    if (n >= NENTITY) n = NENTITY - 1;
    const float4 e = *(const float4*)(ent + (size_t)n * RANK + kq * 4);
    uint2 p;
    p.x = pack2h(e.x, e.y);
    p.y = pack2h(e.z, e.w);
    *(uint2*)(s_ent + rrow * RANK + kq * 4) = p;
  }
  __syncthreads();
#endif

  const int lane  = tid & 31;
  const int wave  = tid >> 5;
  const int bg    = wave & 3;        // batch group (16 rows each)
  const int nhalf = wave >> 2;       // entity half of the tile (64 each)
  const int hs    = (lane >> 4) & 1; // half-wave select (A-layout k split)
  const int lm    = lane & 15;

  // Per-lane pred fragments for this wave's batch group: 4 k-chunks x 2x16B.
  uint4 pf[4][2];
  {
    const uint16_t* pr = s_pred + (bg * 16 + lm) * RANK;
    #pragma unroll
    for (int c = 0; c < 4; ++c) {
      pf[c][0] = *(const uint4*)(pr + c * 32 + hs * 8);
      pf[c][1] = *(const uint4*)(pr + c * 32 + 16 + hs * 8);
    }
  }

  // 4 independent accumulators (one 16x16 tile per entity group)
  v8f acc[4] = {};

  #pragma unroll 1
  for (int jj = 0; jj < 16; ++jj) {  // output column jj: B built once here
    // B = one-hot column jj filled with -1.0h (folds negation + reduction)
    uint32_t bsel = (lm == jj) ? 0xBC00BC00u : 0u;
    AFrag bm;
    #pragma unroll
    for (int r = 0; r < 8; ++r) bm.u[r] = bsel;

    #pragma unroll
    for (int g = 0; g < 4; ++g) {
      const uint16_t* er = s_ent + (nhalf * 64 + g * 16 + jj) * RANK;
      #pragma unroll
      for (int c = 0; c < 4; ++c) {  // k = 128 as 4 chunks of 32
        uint4 e0 = *(const uint4*)(er + c * 32 + hs * 8);       // LDS broadcast
        uint4 e1 = *(const uint4*)(er + c * 32 + 16 + hs * 8);
        AFrag a;
        a.u[0] = absdiff2(pf[c][0].x, e0.x);
        a.u[1] = absdiff2(pf[c][0].y, e0.y);
        a.u[2] = absdiff2(pf[c][0].z, e0.z);
        a.u[3] = absdiff2(pf[c][0].w, e0.w);
        a.u[4] = absdiff2(pf[c][1].x, e1.x);
        a.u[5] = absdiff2(pf[c][1].y, e1.y);
        a.u[6] = absdiff2(pf[c][1].z, e1.z);
        a.u[7] = absdiff2(pf[c][1].w, e1.w);
        // D = A x B + C : k-reduction (and negation) on the matrix pipe
        acc[g] = __builtin_amdgcn_wmma_f32_16x16x32_f16(
            false, a.v, false, bm.v, (short)0, acc[g], false, false);
      }
    }
  }

  // ---- store 4 16x16 tiles: lane -> column n, VGPR r -> row (r + 8*hs) ----
  #pragma unroll
  for (int g = 0; g < 4; ++g) {
    const int ncol = nbase + nhalf * 64 + g * 16 + lm;
    if (ncol < NENTITY) {
      float* op = out + (size_t)(bg * 16 + hs * 8) * NENTITY + ncol;
      #pragma unroll
      for (int r = 0; r < 8; ++r) {
        op[(size_t)r * NENTITY] = acc[g][r];
      }
    }
  }
}

extern "C" void kernel_launch(void* const* d_in, const int* in_sizes, int n_in,
                              void* d_out, int out_size, void* d_ws, size_t ws_size,
                              hipStream_t stream) {
  const float* ent  = (const float*)d_in[0];
  const float* rel  = (const float*)d_in[1];
  const int* heads  = (const int*)d_in[2];
  const int* rels   = (const int*)d_in[3];
  float* out        = (float*)d_out;

  const int grid = (NENTITY + ENT_TILE - 1) / ENT_TILE;  // 391 blocks
  hipLaunchKernelGGL(transe_l1_wmma_kernel, dim3(grid), dim3(256), 0, stream,
                     ent, rel, heads, rels, out);
}
